// QuerySiamese_39891656245252
// MI455X (gfx1250) — matrix operand order repaired and checked by reading, not compiled, not run on hardware
//
#include <hip/hip_runtime.h>
#include <hip/hip_bf16.h>

typedef __attribute__((ext_vector_type(16))) __bf16 v16bf;
typedef __attribute__((ext_vector_type(8)))  __bf16 v8bf;
typedef __attribute__((ext_vector_type(8)))  float  v8f;

#define EMB      768
#define XSTRIDE  (3 * EMB)      // 2304 floats per X row
#define MROWS    32             // rows per wave (two 16-row WMMA blocks)
#define NTILES   600            // 19200 rows / 32
#define H1DIM    256
#define H2DIM    64

// ---------------------------------------------------------------------------
// Weight prep: W [K][N] fp32  ->  Wt [N][K] bf16  (B-operand friendly layout:
// a WMMA B-fragment per lane is then 16 contiguous bf16 = 32 bytes).
// ---------------------------------------------------------------------------
__global__ void transpose_cvt_kernel(const float* __restrict__ W,
                                     __bf16* __restrict__ Wt,
                                     int K, int N) {
    int idx = blockIdx.x * blockDim.x + threadIdx.x;
    if (idx >= K * N) return;
    int k = idx / N;
    int n = idx - k * N;
    Wt[(size_t)n * K + k] = (__bf16)W[(size_t)k * N + n];
}

// ---------------------------------------------------------------------------
// WMMA helper
// ---------------------------------------------------------------------------
__device__ __forceinline__ v8f wmma_bf16(v16bf a, v16bf b, v8f c) {
    return __builtin_amdgcn_wmma_f32_16x16x32_bf16(
        /*neg_a=*/false, a, /*neg_b=*/false, b,
        /*c_mod=*/(short)0, c, /*reuse_a=*/false, /*reuse_b=*/false);
}

// A-fragment (16x32 bf16, contiguous layout): lane = row (m = lane&15), lane half
// selects koff (0 or 8). Elements 0..15 -> K=k0+2*koff+i.
// Two 16-byte ds_load_b128 from a bf16 row in LDS.
__device__ __forceinline__ v16bf load_a_frag_bf16(const __bf16* rowp,
                                                  int k0, int koff) {
    v8bf lo = *(const v8bf*)(rowp + k0 + 2 * koff);
    v8bf hi = *(const v8bf*)(rowp + k0 + 2 * koff + 8);
    v16bf a;
#pragma unroll
    for (int i = 0; i < 8; ++i) { a[i] = lo[i]; a[i + 8] = hi[i]; }
    return a;
}

__device__ __forceinline__ v8bf cvt_pack8(float4 f0, float4 f1) {
    v8bf p;
    p[0] = (__bf16)f0.x; p[1] = (__bf16)f0.y;
    p[2] = (__bf16)f0.z; p[3] = (__bf16)f0.w;
    p[4] = (__bf16)f1.x; p[5] = (__bf16)f1.y;
    p[6] = (__bf16)f1.z; p[7] = (__bf16)f1.w;
    return p;
}

// ---------------------------------------------------------------------------
// Main kernel: one wave (32 lanes) per 32-row tile.
// ---------------------------------------------------------------------------
__global__ void __launch_bounds__(32)
siamese_kernel(const float* __restrict__ X,
               const __bf16* __restrict__ W1t, const __bf16* __restrict__ W3t,
               const __bf16* __restrict__ W2t, const __bf16* __restrict__ W4t,
               const float* __restrict__ b1, const float* __restrict__ b2,
               const float* __restrict__ b3, const float* __restrict__ b4,
               float* __restrict__ out) {
    __shared__ __bf16 Xs[2][MROWS][EMB];    // 96 KB: Xp1 / Xp2 as bf16
    __shared__ __bf16 Hs[3][MROWS][H1DIM];  // 48 KB: H1 / H2 / Hq as bf16

    const int lane = threadIdx.x;
    const int m    = lane & 15;           // row within 16-block (A) / col (B,C)
    const int half = lane >> 4;           // 0: lanes 0-15, 1: lanes 16-31
    const int koff = half * 8;
    const int tile = blockIdx.x;

    // ------------- Stage Xp1/Xp2 rows into LDS as bf16 (read X once) -------
    {
        const float* rp1 = X + ((size_t)tile * MROWS + lane) * XSTRIDE + EMB;
        const float* rp2 = rp1 + EMB;
        for (int c = 0; c < EMB; c += 8) {
            float4 f0 = *(const float4*)(rp1 + c);
            float4 f1 = *(const float4*)(rp1 + c + 4);
            float4 g0 = *(const float4*)(rp2 + c);
            float4 g1 = *(const float4*)(rp2 + c + 4);
            *(v8bf*)(&Xs[0][lane][c]) = cvt_pack8(f0, f1);
            *(v8bf*)(&Xs[1][lane][c]) = cvt_pack8(g0, g1);
        }
    }
    __syncthreads();

    // ---------------- GEMM1: [32x768] x [768x256] for 3 branches ----------
    for (int nc = 0; nc < 8; ++nc) {      // 8 chunks of 2 N-tiles (32 cols)
        v8f acc[3][2][2];                 // [branch][ntile][rowblock]
#pragma unroll
        for (int b = 0; b < 3; ++b)
#pragma unroll
            for (int j = 0; j < 2; ++j)
#pragma unroll
                for (int rb = 0; rb < 2; ++rb)
                    acc[b][j][rb] = v8f{0.f, 0.f, 0.f, 0.f, 0.f, 0.f, 0.f, 0.f};

        for (int k0 = 0; k0 < EMB; k0 += 32) {
            v16bf a1[2], a2[2];
#pragma unroll
            for (int rb = 0; rb < 2; ++rb) {
                a1[rb] = load_a_frag_bf16(&Xs[0][rb * 16 + m][0], k0, koff);
                a2[rb] = load_a_frag_bf16(&Xs[1][rb * 16 + m][0], k0, koff);
            }
#pragma unroll
            for (int j = 0; j < 2; ++j) {
                const int n0 = (nc * 2 + j) * 16;
                const size_t boff = (size_t)(n0 + m) * EMB + k0 + half * 16;
                v16bf bw1 = *(const v16bf*)(W1t + boff);
                v16bf bw3 = *(const v16bf*)(W3t + boff);
#pragma unroll
                for (int rb = 0; rb < 2; ++rb) {
                    acc[0][j][rb] = wmma_bf16(a1[rb], bw1, acc[0][j][rb]);
                    acc[1][j][rb] = wmma_bf16(a2[rb], bw1, acc[1][j][rb]);
                    acc[2][j][rb] = wmma_bf16(a1[rb], bw3, acc[2][j][rb]);
                }
            }
        }

        // bias + ReLU, stash as bf16 in LDS (C layout: lane = col, VGPR = row)
#pragma unroll
        for (int j = 0; j < 2; ++j) {
            const int col = (nc * 2 + j) * 16 + m;
            const float bb1 = b1[col];
            const float bb3 = b3[col];
#pragma unroll
            for (int rb = 0; rb < 2; ++rb) {
#pragma unroll
                for (int r = 0; r < 8; ++r) {
                    const int row = rb * 16 + half * 8 + r;
                    float h1 = acc[0][j][rb][r] + bb1; h1 = h1 > 0.f ? h1 : 0.f;
                    float h2 = acc[1][j][rb][r] + bb1; h2 = h2 > 0.f ? h2 : 0.f;
                    float hq = acc[2][j][rb][r] + bb3; hq = hq > 0.f ? hq : 0.f;
                    Hs[0][row][col] = (__bf16)h1;
                    Hs[1][row][col] = (__bf16)h2;
                    Hs[2][row][col] = (__bf16)hq;
                }
            }
        }
    }
    __syncthreads();   // single-wave WG: lowers to LDS waitcnt ordering

    // ---------------- GEMM2: [32x256] x [256x64] + reduction ---------------
    float accq[2][8], accd[2][8];
#pragma unroll
    for (int rb = 0; rb < 2; ++rb)
#pragma unroll
        for (int r = 0; r < 8; ++r) { accq[rb][r] = 0.f; accd[rb][r] = 0.f; }

    for (int j2 = 0; j2 < 4; ++j2) {      // 4 N-tiles of 16 cols = 64
        v8f z1[2], z2[2], zq[2];
#pragma unroll
        for (int rb = 0; rb < 2; ++rb) {
            z1[rb] = v8f{0.f, 0.f, 0.f, 0.f, 0.f, 0.f, 0.f, 0.f};
            z2[rb] = z1[rb]; zq[rb] = z1[rb];
        }
        for (int k0 = 0; k0 < H1DIM; k0 += 32) {
            const size_t boff = (size_t)(j2 * 16 + m) * H1DIM + k0 + half * 16;
            v16bf w2f = *(const v16bf*)(W2t + boff);
            v16bf w4f = *(const v16bf*)(W4t + boff);
#pragma unroll
            for (int rb = 0; rb < 2; ++rb) {
                v16bf h1f = load_a_frag_bf16(&Hs[0][rb * 16 + m][0], k0, koff);
                v16bf h2f = load_a_frag_bf16(&Hs[1][rb * 16 + m][0], k0, koff);
                v16bf hqf = load_a_frag_bf16(&Hs[2][rb * 16 + m][0], k0, koff);
                z1[rb] = wmma_bf16(h1f, w2f, z1[rb]);
                z2[rb] = wmma_bf16(h2f, w2f, z2[rb]);
                zq[rb] = wmma_bf16(hqf, w4f, zq[rb]);
            }
        }
        const int col = j2 * 16 + m;
        const float bb2 = b2[col];
        const float bb4 = b4[col];
#pragma unroll
        for (int rb = 0; rb < 2; ++rb)
#pragma unroll
            for (int r = 0; r < 8; ++r) {
                float a1 = z1[rb][r] + bb2; a1 = a1 > 0.f ? a1 : 0.f;
                float a2 = z2[rb][r] + bb2; a2 = a2 > 0.f ? a2 : 0.f;
                float aq = zq[rb][r] + bb4; aq = aq > 0.f ? aq : 0.f;
                const float d = a1 - a2;
                accd[rb][r] += d * d;
                accq[rb][r] += aq * aq;
            }
    }

    // Cross-lane sum over the 16 columns held by each lane-half.
#pragma unroll
    for (int rb = 0; rb < 2; ++rb)
#pragma unroll
        for (int r = 0; r < 8; ++r) {
#pragma unroll
            for (int mask = 1; mask <= 8; mask <<= 1) {
                accq[rb][r] += __shfl_xor(accq[rb][r], mask, 32);
                accd[rb][r] += __shfl_xor(accd[rb][r], mask, 32);
            }
        }

    if (m == 0) {                         // lanes 0 and 16 write their half
#pragma unroll
        for (int rb = 0; rb < 2; ++rb)
#pragma unroll
            for (int r = 0; r < 8; ++r) {
                const int row = tile * MROWS + rb * 16 + half * 8 + r;
                out[row] = accq[rb][r] * accd[rb][r] * (1.0f / 4096.0f);
            }
    }
}

// ---------------------------------------------------------------------------
// Launch
// ---------------------------------------------------------------------------
extern "C" void kernel_launch(void* const* d_in, const int* in_sizes, int n_in,
                              void* d_out, int out_size, void* d_ws, size_t ws_size,
                              hipStream_t stream) {
    const float* X  = (const float*)d_in[0];
    const float* W1 = (const float*)d_in[1];
    const float* b1 = (const float*)d_in[2];
    const float* W2 = (const float*)d_in[3];
    const float* b2 = (const float*)d_in[4];
    const float* W3 = (const float*)d_in[5];
    const float* b3 = (const float*)d_in[6];
    const float* W4 = (const float*)d_in[7];
    const float* b4 = (const float*)d_in[8];
    float* out = (float*)d_out;

    __bf16* W1t = (__bf16*)d_ws;                 // [256][768]
    __bf16* W3t = W1t + (size_t)256 * 768;       // [256][768]
    __bf16* W2t = W3t + (size_t)256 * 768;       // [64][256]
    __bf16* W4t = W2t + (size_t)64 * 256;        // [64][256]

    transpose_cvt_kernel<<<(768 * 256 + 255) / 256, 256, 0, stream>>>(W1, W1t, 768, 256);
    transpose_cvt_kernel<<<(768 * 256 + 255) / 256, 256, 0, stream>>>(W3, W3t, 768, 256);
    transpose_cvt_kernel<<<(256 * 64 + 255) / 256, 256, 0, stream>>>(W2, W2t, 256, 64);
    transpose_cvt_kernel<<<(256 * 64 + 255) / 256, 256, 0, stream>>>(W4, W4t, 256, 64);

    siamese_kernel<<<NTILES, 32, 0, stream>>>(X, W1t, W3t, W2t, W4t,
                                              b1, b2, b3, b4, out);
}
